// DIN_82429012345213
// MI455X (gfx1250) — matrix operand order repaired and checked by reading, not compiled, not run on hardware
//
#include <hip/hip_runtime.h>
#include <math.h>

typedef __attribute__((ext_vector_type(16))) _Float16 v16h;
typedef __attribute__((ext_vector_type(8)))  float    v8f;

union V16 { v16h v; uint4 u[2]; };

#define Bsz 1024
#define Lseq 200
#define DIT 96
#define KTOT 192          // [seq | seq*cand] folded K
#define MPAD 208          // 13 tiles of 16
#define LDST 208          // LDS stride in halves (416B, 16B aligned)
#define MT_N 13
#define NT_N 6
#define KK_N 6            // 192 / 32

// ---- dynamic LDS layout for att_kernel (bytes) ----
constexpr int ATT_SMEM = (MPAD * LDST + 96 * LDST) * 2 + (208 + 208 + 256 + 96 + 96 + 96) * 4 + 400 * 4;

// K0: fold att_w1 (384x96 row-major) into transposed f16 Wt[n*192+k]
//   k<96 : W0+W2 (seq coeff), k>=96 : W3 (seq*t coeff). row(192+k) covers both.
__global__ __launch_bounds__(256) void fold_kernel(const float* __restrict__ w1,
                                                   _Float16* __restrict__ Wt) {
  int idx = blockIdx.x * 256 + threadIdx.x;
  if (idx >= 96 * KTOT) return;
  int n = idx / KTOT, k = idx - n * KTOT;
  float v = w1[(192 + k) * 96 + n];
  if (k < 96) v += w1[k * 96 + n];
  Wt[idx] = (_Float16)v;
}

// K1: base[b][j] = att_b1[j] + sum_i cand[i] * (w1[96+i][j] - w1[192+i][j])
__global__ __launch_bounds__(128) void base_kernel(
    const int* __restrict__ cand_id, const int* __restrict__ cand_cat,
    const float* __restrict__ emb_id, const float* __restrict__ emb_cat,
    const float* __restrict__ w1, const float* __restrict__ b1,
    float* __restrict__ basep) {
  __shared__ float c[96];
  int b = blockIdx.x, tid = threadIdx.x;
  if (tid < 96)
    c[tid] = (tid < 64) ? emb_id[cand_id[b] * 64 + tid]
                        : emb_cat[cand_cat[b] * 32 + (tid - 64)];
  __syncthreads();
  if (tid < 96) {
    float acc = b1[tid];
    for (int i = 0; i < 96; ++i)
      acc += c[i] * (w1[(96 + i) * 96 + tid] - w1[(192 + i) * 96 + tid]);
    basep[b * 96 + tid] = acc;
  }
}

// K2: fused attention unit per (b, sequence). WMMA f16 GEMM 208x96x192 in LDS.
__global__ __launch_bounds__(256) void att_kernel(
    const int* __restrict__ exp_id, const int* __restrict__ exp_cat,
    const int* __restrict__ ipv_id, const int* __restrict__ ipv_cat,
    const int* __restrict__ cand_id, const int* __restrict__ cand_cat,
    const float* __restrict__ emb_id, const float* __restrict__ emb_cat,
    const _Float16* __restrict__ Wt, const float* __restrict__ basep,
    const float* __restrict__ w2, float* __restrict__ attn_out) {
  extern __shared__ __align__(16) char smem[];
  _Float16* sA = (_Float16*)smem;                 // [MPAD][LDST]
  _Float16* sW = sA + MPAD * LDST;                // [96][LDST]
  float* sLogit  = (float*)(smem + (MPAD * LDST + 96 * LDST) * 2);
  float* sWeight = sLogit + 208;
  float* sRed    = sWeight + 208;                 // 256 floats (multi-purpose)
  float* sCand   = sRed + 256;
  float* sBase   = sCand + 96;
  float* sW2     = sBase + 96;
  int* sIdxId    = (int*)(sW2 + 96);
  int* sIdxCat   = sIdxId + 200;

  const int b = blockIdx.x >> 1;
  const int sidx = blockIdx.x & 1;
  const int tid = threadIdx.x;
  const int* idp = sidx ? ipv_id : exp_id;
  const int* icp = sidx ? ipv_cat : exp_cat;

  if (tid < Lseq) { sIdxId[tid] = idp[b * Lseq + tid]; sIdxCat[tid] = icp[b * Lseq + tid]; }
  if (tid < 96) {
    sCand[tid] = (tid < 64) ? emb_id[cand_id[b] * 64 + tid]
                            : emb_cat[cand_cat[b] * 32 + (tid - 64)];
    sBase[tid] = basep[b * 96 + tid];
    sW2[tid]   = w2[tid];
  }
  if (tid < 208) sLogit[tid] = 0.f;
  __syncthreads();

  // Stage A = [seq | seq*cand] as f16
  for (int idx = tid; idx < Lseq * 96; idx += 256) {
    int l = idx / 96, j = idx - l * 96;
    float v = (j < 64) ? emb_id[sIdxId[l] * 64 + j]
                       : emb_cat[sIdxCat[l] * 32 + (j - 64)];
    sA[l * LDST + j]      = (_Float16)v;
    sA[l * LDST + 96 + j] = (_Float16)(v * sCand[j]);
  }
  for (int idx = tid; idx < 8 * KTOT; idx += 256) {       // zero pad rows 200..207
    int l = Lseq + idx / KTOT, j = idx - (idx / KTOT) * KTOT;
    sA[l * LDST + j] = (_Float16)0.f;
  }
  for (int idx = tid; idx < 96 * KTOT; idx += 256) {      // stage Wt transposed [n][k]
    int n = idx / KTOT, k = idx - n * KTOT;
    sW[n * LDST + k] = Wt[idx];
  }
  __syncthreads();

  const int wave = tid >> 5;
  const int lane = tid & 31;
  const int g = lane >> 4;        // lane half-group
  const int ln = lane & 15;       // M (for A) / N (for B/C)

  for (int t = wave; t < MT_N * NT_N; t += 8) {
    int mt = t % MT_N, nt = t / MT_N;
    v8f acc = {};
    // A frag: row = mt*16+ln, chunks at K = 32kk+8g (+16 halves for VGPR4-7)
    const char* arow = (const char*)(sA + (mt * 16 + ln) * LDST) + 16 * g;
    // B frag: row n = nt*16+ln, chunks at K = 32kk+8g (+16 halves for VGPR4-7)
    const char* brow = (const char*)(sW + (nt * 16 + ln) * LDST) + 16 * g;
#pragma unroll
    for (int kk = 0; kk < KK_N; ++kk) {
      V16 a, bb;
      a.u[0]  = *(const uint4*)(arow + 64 * kk);
      a.u[1]  = *(const uint4*)(arow + 64 * kk + 32);
      bb.u[0] = *(const uint4*)(brow + 64 * kk);
      bb.u[1] = *(const uint4*)(brow + 64 * kk + 32);
      acc = __builtin_amdgcn_wmma_f32_16x16x32_f16(
          false, a.v, false, bb.v, (short)0, acc, false, false);
    }
    // Fused epilogue: +base, ReLU, partial h@w2 into sLogit.
    // Rows >= 200 only occur for mt==12, g==1: contribute 0 instead of
    // branching so EXEC stays full (no divergent atomics, no saveexec chains).
    int n = nt * 16 + ln;
    float bn = sBase[n];
    float w2eff = ((mt < 12) | (g == 0)) ? sW2[n] : 0.f;
#pragma unroll
    for (int r = 0; r < 8; ++r) {
      int m = mt * 16 + r + 8 * g;   // C layout: VGPR r -> row r+8g; m <= 207
      float hv = acc[r] + bn;
      hv = hv > 0.f ? hv : 0.f;
      atomicAdd(&sLogit[m], hv * w2eff);
    }
  }
  __syncthreads();

  // Softmax over L: wave32 butterfly reductions + 8-wave combine
  float lv = (tid < Lseq) ? sLogit[tid] : -3.0e38f;
  float mxl = lv;
#pragma unroll
  for (int off = 16; off > 0; off >>= 1)
    mxl = fmaxf(mxl, __shfl_xor(mxl, off, 32));
  if (lane == 0) sRed[wave] = mxl;
  __syncthreads();
  float mx = sRed[0];
#pragma unroll
  for (int i = 1; i < 8; ++i) mx = fmaxf(mx, sRed[i]);

  float e = 0.f;
  if (tid < Lseq) { e = expf(lv - mx); sWeight[tid] = e; }
  float sm = e;
#pragma unroll
  for (int off = 16; off > 0; off >>= 1)
    sm += __shfl_xor(sm, off, 32);
  if (lane == 0) sRed[8 + wave] = sm;
  __syncthreads();
  float tot = 0.f;
#pragma unroll
  for (int i = 0; i < 8; ++i) tot += sRed[8 + i];
  float inv = 1.f / tot;
  __syncthreads();   // done reading sRed[8..15]; reuse sRed as partial buffer

  // Weighted sum of seq (f16 S columns 0..95), L split in two halves
  if (tid < 192) {
    int j = tid % 96, h = tid / 96;
    float acc = 0.f;
    for (int l = 100 * h; l < 100 * h + 100; ++l)
      acc += sWeight[l] * (float)sA[l * LDST + j];
    sRed[tid] = acc;
  }
  __syncthreads();
  if (tid < 96)
    attn_out[(b * 2 + sidx) * 96 + tid] = (sRed[tid] + sRed[tid + 96]) * inv;
}

// K3: concat + MLP + BN(eval) + sigmoid(mlp + logit(score))
__global__ __launch_bounds__(256) void mlp_kernel(
    const float* __restrict__ num1, const float* __restrict__ num2,
    const float* __restrict__ score, const int* __restrict__ user_id,
    const int* __restrict__ cand_id, const int* __restrict__ cand_cat,
    const float* __restrict__ emb_user, const float* __restrict__ emb_id,
    const float* __restrict__ emb_cat, const float* __restrict__ attn,
    const float* __restrict__ fc1w, const float* __restrict__ fc1b,
    const float* __restrict__ g1, const float* __restrict__ be1,
    const float* __restrict__ m1, const float* __restrict__ v1,
    const float* __restrict__ fc2w, const float* __restrict__ fc2b,
    const float* __restrict__ g2, const float* __restrict__ be2,
    const float* __restrict__ m2, const float* __restrict__ v2,
    const float* __restrict__ fc3w, const float* __restrict__ fc3b,
    float* __restrict__ out) {
  __shared__ float x[322];
  __shared__ float h1[200];
  __shared__ float h2[80];
  int b = blockIdx.x, tid = threadIdx.x;
  if (tid == 0) { x[0] = num1[b]; x[1] = num2[b]; }
  if (tid < 32) x[2 + tid]  = emb_user[user_id[b] * 32 + tid];
  if (tid < 64) x[34 + tid] = emb_id[cand_id[b] * 64 + tid];
  if (tid < 32) x[98 + tid] = emb_cat[cand_cat[b] * 32 + tid];
  if (tid < 96) {
    x[130 + tid] = attn[(b * 2 + 0) * 96 + tid];
    x[226 + tid] = attn[(b * 2 + 1) * 96 + tid];
  }
  __syncthreads();
  if (tid < 200) {
    float acc = fc1b[tid];
    for (int i = 0; i < 322; ++i) acc += x[i] * fc1w[i * 200 + tid];
    acc = (acc - m1[tid]) * rsqrtf(v1[tid] + 1e-5f) * g1[tid] + be1[tid];
    h1[tid] = acc > 0.f ? acc : 0.f;
  }
  __syncthreads();
  if (tid < 80) {
    float acc = fc2b[tid];
    for (int i = 0; i < 200; ++i) acc += h1[i] * fc2w[i * 80 + tid];
    acc = (acc - m2[tid]) * rsqrtf(v2[tid] + 1e-5f) * g2[tid] + be2[tid];
    h2[tid] = acc > 0.f ? acc : 0.f;
  }
  __syncthreads();
  if (tid == 0) {
    float acc = fc3b[0];
    for (int i = 0; i < 80; ++i) acc += h2[i] * fc3w[i];
    float s = score[b];
    float z = acc + logf(s / (1.f - s));
    out[b] = 1.f / (1.f + expf(-z));
  }
}

extern "C" void kernel_launch(void* const* d_in, const int* in_sizes, int n_in,
                              void* d_out, int out_size, void* d_ws, size_t ws_size,
                              hipStream_t stream) {
  const float* num1     = (const float*)d_in[0];
  const float* num2     = (const float*)d_in[1];
  const float* score    = (const float*)d_in[2];
  const int*   user_id  = (const int*)d_in[3];
  const int*   cand_id  = (const int*)d_in[4];
  const int*   cand_cat = (const int*)d_in[5];
  const int*   exp_id   = (const int*)d_in[6];
  const int*   exp_cat  = (const int*)d_in[7];
  const int*   ipv_id   = (const int*)d_in[8];
  const int*   ipv_cat  = (const int*)d_in[9];
  const float* emb_id   = (const float*)d_in[10];
  const float* emb_cat  = (const float*)d_in[11];
  const float* emb_user = (const float*)d_in[12];
  const float* att_w1   = (const float*)d_in[13];
  const float* att_b1   = (const float*)d_in[14];
  const float* att_w2   = (const float*)d_in[15];
  /* att_b2 (d_in[16]) cancels in softmax */
  const float* fc1w = (const float*)d_in[17];
  const float* fc1b = (const float*)d_in[18];
  const float* g1   = (const float*)d_in[19];
  const float* be1  = (const float*)d_in[20];
  const float* m1   = (const float*)d_in[21];
  const float* v1   = (const float*)d_in[22];
  const float* fc2w = (const float*)d_in[23];
  const float* fc2b = (const float*)d_in[24];
  const float* g2   = (const float*)d_in[25];
  const float* be2  = (const float*)d_in[26];
  const float* m2   = (const float*)d_in[27];
  const float* v2   = (const float*)d_in[28];
  const float* fc3w = (const float*)d_in[29];
  const float* fc3b = (const float*)d_in[30];
  float* out = (float*)d_out;

  // Workspace carve-up
  char* ws = (char*)d_ws;
  _Float16* Wt = (_Float16*)ws;                              // 96*192*2   = 36864 B
  float* basep = (float*)(ws + 36864);                       // 1024*96*4  = 393216 B
  float* attn  = (float*)(ws + 36864 + 393216);              // 1024*2*96*4= 786432 B

  (void)hipFuncSetAttribute(reinterpret_cast<const void*>(att_kernel),
                            hipFuncAttributeMaxDynamicSharedMemorySize, ATT_SMEM);

  fold_kernel<<<(96 * KTOT + 255) / 256, 256, 0, stream>>>(att_w1, Wt);
  base_kernel<<<Bsz, 128, 0, stream>>>(cand_id, cand_cat, emb_id, emb_cat,
                                       att_w1, att_b1, basep);
  att_kernel<<<Bsz * 2, 256, ATT_SMEM, stream>>>(
      exp_id, exp_cat, ipv_id, ipv_cat, cand_id, cand_cat,
      emb_id, emb_cat, Wt, basep, att_w2, attn);
  mlp_kernel<<<Bsz, 256, 0, stream>>>(
      num1, num2, score, user_id, cand_id, cand_cat, emb_user, emb_id, emb_cat,
      attn, fc1w, fc1b, g1, be1, m1, v1, fc2w, fc2b, g2, be2, m2, v2,
      fc3w, fc3b, out);
}
